// minGRUCell_87230785782007
// MI455X (gfx1250) — compile-verified, matrix operand order checked by reading, not run on hardware
//
#include <hip/hip_runtime.h>
#include <math.h>

typedef __attribute__((ext_vector_type(16))) _Float16 v16h;
typedef __attribute__((ext_vector_type(8)))  _Float16 v8h;
typedef __attribute__((ext_vector_type(8)))  float    v8f;

#define B_    4
#define L_    2048
#define DIM_  1024
#define KCONV 4
#define EXP_  1536
#define ROWS  (B_ * L_)          // 8192 token rows

// GEMM tiling: 256 thr = 8 waves (4 in M x 2 in N), 64x64 per wave -> 256x128 block
#define BLK_M 256
#define BLK_N 128
#define BLK_K 32

// ---------------------------------------------------------------------------
// f32 -> f16 conversion (weights, once per launch; deterministic)
// ---------------------------------------------------------------------------
__global__ __launch_bounds__(256) void cvt_f32_f16_kernel(
    const float* __restrict__ in, _Float16* __restrict__ out, int n) {
  int i = blockIdx.x * 256 + threadIdx.x;
  if (i < n) out[i] = (_Float16)in[i];
}

// ---------------------------------------------------------------------------
// LayerNorm over DIM=1024, one block (256 thr) per row, writes f16
// ---------------------------------------------------------------------------
__global__ __launch_bounds__(256) void ln_f16_kernel(
    const float* __restrict__ x, const float* __restrict__ g,
    const float* __restrict__ b, _Float16* __restrict__ out) {
  __shared__ float red[256];
  const int row = blockIdx.x;
  const float* xr = x + (size_t)row * DIM_;
  float v[4];
  float s = 0.f;
#pragma unroll
  for (int i = 0; i < 4; ++i) { v[i] = xr[threadIdx.x + i * 256]; s += v[i]; }
  red[threadIdx.x] = s;
  __syncthreads();
  for (int off = 128; off > 0; off >>= 1) {
    if (threadIdx.x < off) red[threadIdx.x] += red[threadIdx.x + off];
    __syncthreads();
  }
  const float mean = red[0] * (1.f / DIM_);
  __syncthreads();
  float s2 = 0.f;
#pragma unroll
  for (int i = 0; i < 4; ++i) { float d = v[i] - mean; s2 += d * d; }
  red[threadIdx.x] = s2;
  __syncthreads();
  for (int off = 128; off > 0; off >>= 1) {
    if (threadIdx.x < off) red[threadIdx.x] += red[threadIdx.x + off];
    __syncthreads();
  }
  const float rstd = rsqrtf(red[0] * (1.f / DIM_) + 1e-5f);
#pragma unroll
  for (int i = 0; i < 4; ++i) {
    const int d = threadIdx.x + i * 256;
    out[(size_t)row * DIM_ + d] = (_Float16)((v[i] - mean) * rstd * g[d] + b[d]);
  }
}

// ---------------------------------------------------------------------------
// Causal depthwise conv1d (K=4), per-element, f16 in -> f16 out
// ---------------------------------------------------------------------------
__global__ __launch_bounds__(256) void dwconv_kernel(
    const _Float16* __restrict__ in, const float* __restrict__ w,
    const float* __restrict__ bias, _Float16* __restrict__ out) {
  const int idx = blockIdx.x * 256 + threadIdx.x;   // over B*L*DIM
  const int d   = idx & (DIM_ - 1);
  const int bl  = idx >> 10;                         // b*L + l
  const int l   = bl & (L_ - 1);
  float acc = bias[d];
#pragma unroll
  for (int t = 0; t < KCONV; ++t) {
    const int ll = l - (KCONV - 1) + t;
    if (ll >= 0)
      acc += w[d * KCONV + t] * (float)in[(size_t)(bl - (KCONV - 1) + t) * DIM_ + d];
  }
  out[(size_t)bl * DIM_ + d] = (_Float16)acc;
}

// ---------------------------------------------------------------------------
// minGRU recurrence: h_t = sigmoid(-k)*h_{t-1} + sigmoid(k)*g(hx), h_0=0.5
// (positive-domain equivalent of the reference log-space parallel scan)
// ---------------------------------------------------------------------------
__global__ __launch_bounds__(256) void mingru_scan_kernel(
    const float* __restrict__ kh, _Float16* __restrict__ hseq16) {
  const int tid = blockIdx.x * 256 + threadIdx.x;    // over B*EXP
  const int e = tid % EXP_;
  const int b = tid / EXP_;
  const float* base = kh + (size_t)b * L_ * (2 * EXP_);
  float h = 0.5f;
  for (int l = 0; l < L_; ++l) {
    const float k  = base[(size_t)l * (2 * EXP_) + e];
    const float hx = base[(size_t)l * (2 * EXP_) + EXP_ + e];
    const float z  = 1.f / (1.f + __expf(-k));       // sigmoid(k)
    const float gh = (hx >= 0.f) ? (hx + 0.5f) : (1.f / (1.f + __expf(-hx)));
    h = (1.f - z) * h + z * gh;
    hseq16[((size_t)b * L_ + l) * EXP_ + e] = (_Float16)h;
  }
}

// ---------------------------------------------------------------------------
// WMMA GEMM: C[M,N] = A[M,K] (f16 row-major) x W[N,K]^T (f16 row-major)
//  - B (weight) tile staged in LDS via global_load_async_to_lds_b128,
//    double-buffered, synced with s_wait_asynccnt + workgroup barrier
//  - per-wave 64x64 tile = 4x4 v_wmma_f32_16x16x32_f16 accumulators
//  - epilogue fuses +bias, exact GELU, +residual, f32/f16 stores
// ---------------------------------------------------------------------------
__device__ __forceinline__ v16h frag_from(const _Float16* p) {
  // 16-bit A/B fragment: lane<16 -> K {0..7,16..23}; lane>=16 -> {8..15,24..31}
  v8h lo = *(const v8h*)(p);
  v8h hi = *(const v8h*)(p + 16);
  return __builtin_shufflevector(lo, hi, 0, 1, 2, 3, 4, 5, 6, 7,
                                 8, 9, 10, 11, 12, 13, 14, 15);
}

__device__ __forceinline__ void epi_store(float v, int row, int col, int N,
                                          const float* __restrict__ bias,
                                          const float* __restrict__ resid,
                                          float* __restrict__ outF,
                                          _Float16* __restrict__ outH,
                                          int actGelu) {
  if (bias) v += bias[col];
  if (actGelu) v = 0.5f * v * (1.f + erff(v * 0.70710678118654752f));
  if (resid) v += resid[(size_t)row * N + col];
  const size_t idx = (size_t)row * N + col;
  if (outF) outF[idx] = v;
  if (outH) outH[idx] = (_Float16)v;
}

__global__ __launch_bounds__(256) void gemm_wmma_kernel(
    const _Float16* __restrict__ A, const _Float16* __restrict__ W,
    const float* __restrict__ bias, const float* __restrict__ resid,
    float* __restrict__ outF, _Float16* __restrict__ outH,
    int N, int K, int actGelu) {
  __shared__ alignas(16) _Float16 Bsh[2][BLK_N * BLK_K];   // 2 x 8KB

  const int lane = threadIdx.x & 31;
  const int w    = threadIdx.x >> 5;
  const int m0   = blockIdx.x * BLK_M + (w & 3) * 64;
  const int wn   = w >> 2;                                  // 0..1
  const int n0   = blockIdx.y * BLK_N + wn * 64;
  const int mr   = lane & 15;
  const int khalf = (lane < 16) ? 0 : 8;

  // async copy of W[nbase .. nbase+127][kk .. kk+31] into Bsh[buf]
  const int nbase = blockIdx.y * BLK_N;
  const int c0    = (int)threadIdx.x;                       // chunk ids c0, c0+256
  auto issue_b_tile = [&](int buf, int kk) {
#pragma unroll
    for (int c = 0; c < 2; ++c) {
      const int chunk = c0 + c * 256;                       // 512 x 16B chunks
      const int row = chunk >> 2;
      const int ko  = (chunk & 3) * 8;
      const _Float16* g = W + (size_t)(nbase + row) * K + kk + ko;
      const unsigned lds = (unsigned)(size_t)(&Bsh[buf][row * BLK_K + ko]);
      asm volatile("global_load_async_to_lds_b128 %0, %1, off"
                   :: "v"(lds), "v"(g) : "memory");
    }
  };

  // hoisted per-lane A row pointers (advance by k0 inside the loop)
  const _Float16* pA[4];
#pragma unroll
  for (int mt = 0; mt < 4; ++mt)
    pA[mt] = A + (size_t)(m0 + mt * 16 + mr) * K + khalf;
  // per-lane B fragment base offsets inside the LDS tile
  int bOff[4];
#pragma unroll
  for (int nt = 0; nt < 4; ++nt)
    bOff[nt] = (wn * 64 + nt * 16 + mr) * BLK_K + khalf;

  v8f acc[4][4];
#pragma unroll
  for (int mt = 0; mt < 4; ++mt)
#pragma unroll
    for (int nt = 0; nt < 4; ++nt) acc[mt][nt] = v8f{};

  // pipeline prologue: stage first B tile
  issue_b_tile(0, 0);
  asm volatile("s_wait_asynccnt 0x0" ::: "memory");
  __syncthreads();

  int buf = 0;
  for (int k0 = 0; k0 < K; k0 += BLK_K) {
    if (k0 + BLK_K < K) issue_b_tile(buf ^ 1, k0 + BLK_K);  // overlap next tile

    v16h af[4], bf[4];
#pragma unroll
    for (int mt = 0; mt < 4; ++mt) af[mt] = frag_from(pA[mt] + k0);
#pragma unroll
    for (int nt = 0; nt < 4; ++nt) bf[nt] = frag_from(&Bsh[buf][bOff[nt]]);

#pragma unroll
    for (int mt = 0; mt < 4; ++mt)
#pragma unroll
      for (int nt = 0; nt < 4; ++nt)
        acc[mt][nt] = __builtin_amdgcn_wmma_f32_16x16x32_f16(
            false, af[mt], false, bf[nt], (short)0, acc[mt][nt], false, false);

    if (k0 + BLK_K < K)
      asm volatile("s_wait_asynccnt 0x0" ::: "memory");
    __syncthreads();
    buf ^= 1;
  }

  // C/D layout: lane<16 -> (M=r, N=lane); lane>=16 -> (M=8+r, N=lane-16)
  const int col0 = n0 + mr;
  const int roff = (lane < 16) ? 0 : 8;
#pragma unroll
  for (int mt = 0; mt < 4; ++mt)
#pragma unroll
    for (int nt = 0; nt < 4; ++nt)
#pragma unroll
      for (int r = 0; r < 8; ++r)
        epi_store(acc[mt][nt][r], m0 + mt * 16 + roff + r, col0 + nt * 16,
                  N, bias, resid, outF, outH, actGelu);
}

// ---------------------------------------------------------------------------
// Host-side orchestration
// ---------------------------------------------------------------------------
static inline size_t align256(size_t x) { return (x + 255) & ~(size_t)255; }

extern "C" void kernel_launch(void* const* d_in, const int* in_sizes, int n_in,
                              void* d_out, int out_size, void* d_ws, size_t ws_size,
                              hipStream_t stream) {
  const float* x     = (const float*)d_in[0];
  const float* dw_w  = (const float*)d_in[1];
  const float* dw_b  = (const float*)d_in[2];
  const float* pw_w  = (const float*)d_in[3];
  const float* pw_b  = (const float*)d_in[4];
  const float* ln1_g = (const float*)d_in[5];
  const float* ln1_b = (const float*)d_in[6];
  const float* f_w   = (const float*)d_in[7];
  const float* f_b   = (const float*)d_in[8];
  const float* dp_w  = (const float*)d_in[9];
  const float* ln2_g = (const float*)d_in[10];
  const float* ln2_b = (const float*)d_in[11];
  const float* mlp_w1 = (const float*)d_in[12];
  const float* mlp_b1 = (const float*)d_in[13];
  const float* mlp_w2 = (const float*)d_in[14];
  const float* mlp_b2 = (const float*)d_in[15];
  const float* ln3_g = (const float*)d_in[16];
  const float* ln3_b = (const float*)d_in[17];
  float* out = (float*)d_out;

  // ---- workspace layout ----
  char* ws = (char*)d_ws;
  size_t off = 0;
  _Float16* pw16 = ( _Float16*)(ws + off); off = align256(off + (size_t)DIM_ * DIM_ * 2);
  _Float16* fw16 = ( _Float16*)(ws + off); off = align256(off + (size_t)2 * EXP_ * DIM_ * 2);
  _Float16* dp16 = ( _Float16*)(ws + off); off = align256(off + (size_t)DIM_ * EXP_ * 2);
  _Float16* w116 = ( _Float16*)(ws + off); off = align256(off + (size_t)4 * DIM_ * DIM_ * 2);
  _Float16* w216 = ( _Float16*)(ws + off); off = align256(off + (size_t)4 * DIM_ * DIM_ * 2);
  _Float16* actA = ( _Float16*)(ws + off); off = align256(off + (size_t)ROWS * DIM_ * 2);   // LN outs
  _Float16* actB = ( _Float16*)(ws + off); off = align256(off + (size_t)ROWS * EXP_ * 2);   // conv16 / hseq16
  float*    big  = (float*)(ws + off);     off = align256(off + (size_t)ROWS * 2 * EXP_ * 4); // kh; reused as gelu16
  float*    x1   = (float*)(ws + off);     off = align256(off + (size_t)ROWS * DIM_ * 4);
  float*    x2   = (float*)(ws + off);     off = align256(off + (size_t)ROWS * DIM_ * 4);
  _Float16* gelu16 = (_Float16*)big;  // lifetime-disjoint with kh

  // ---- weight conversion (f32 -> f16) ----
  auto cvt = [&](const float* src, _Float16* dst, int n) {
    cvt_f32_f16_kernel<<<(n + 255) / 256, 256, 0, stream>>>(src, dst, n);
  };
  cvt(pw_w, pw16, DIM_ * DIM_);
  cvt(f_w, fw16, 2 * EXP_ * DIM_);
  cvt(dp_w, dp16, DIM_ * EXP_);
  cvt(mlp_w1, w116, 4 * DIM_ * DIM_);
  cvt(mlp_w2, w216, 4 * DIM_ * DIM_);

  const dim3 blk(256);
  const dim3 gridN1024(ROWS / BLK_M, 1024 / BLK_N);
  const dim3 gridN3072(ROWS / BLK_M, 3072 / BLK_N);
  const dim3 gridN4096(ROWS / BLK_M, 4096 / BLK_N);

  // ---- block 1: LN1 -> dwconv -> pointwise GEMM (+bias, +residual x) ----
  ln_f16_kernel<<<ROWS, blk, 0, stream>>>(x, ln1_g, ln1_b, actA);
  dwconv_kernel<<<(ROWS * DIM_) / 256, blk, 0, stream>>>(actA, dw_w, dw_b, actB);
  gemm_wmma_kernel<<<gridN1024, blk, 0, stream>>>(actB, pw16, pw_b, x, x1, nullptr,
                                                  DIM_, DIM_, 0);

  // ---- block 2: LN2 -> f GEMM -> minGRU scan -> down-proj (+residual x1) ----
  ln_f16_kernel<<<ROWS, blk, 0, stream>>>(x1, ln2_g, ln2_b, actA);
  gemm_wmma_kernel<<<gridN3072, blk, 0, stream>>>(actA, fw16, f_b, nullptr, big, nullptr,
                                                  2 * EXP_, DIM_, 0);
  mingru_scan_kernel<<<(B_ * EXP_) / 256, blk, 0, stream>>>(big, actB);
  gemm_wmma_kernel<<<gridN1024, blk, 0, stream>>>(actB, dp16, nullptr, x1, x2, nullptr,
                                                  DIM_, EXP_, 0);

  // ---- block 3: LN3 -> MLP up (+bias, GELU) -> MLP down (+bias, +residual x2) ----
  ln_f16_kernel<<<ROWS, blk, 0, stream>>>(x2, ln3_g, ln3_b, actA);
  gemm_wmma_kernel<<<gridN4096, blk, 0, stream>>>(actA, w116, mlp_b1, nullptr,
                                                  nullptr, gelu16, 4 * DIM_, DIM_, 1);
  gemm_wmma_kernel<<<gridN1024, blk, 0, stream>>>(gelu16, w216, mlp_b2, x2, out, nullptr,
                                                  DIM_, 4 * DIM_, 0);
  (void)in_sizes; (void)n_in; (void)out_size; (void)ws_size;
}
